// SISW_28939489640823
// MI455X (gfx1250) — matrix operand order repaired and checked by reading, not compiled, not run on hardware
//
#include <hip/hip_runtime.h>

// ---------------- problem constants ----------------
#define HWp   40000
#define Wd    200
#define Hd    200
#define Cc    64
#define NIMG  15          // 3 tg + 12 nb
#define NS    12          // BT*(N-1)
#define EPSf  1e-8f

// ---------------- workspace layout (float offsets) ----------------
#define OFF_COMP   0            // 15*HW   compressed+relu
#define OFF_ENT    600000       // 15*HW   entropy maps
#define OFF_MASK   1200000      // 12*HW   mask
#define OFF_DEN    1680000      // 12*HW   box3(mask)
#define OFF_WARP   2160000      // 12*64*HW warp feats
#define OFF_LOGIT  32880000     // 15*HW   pwf logits
#define OFF_BW     33480000     // 1       mask sum
#define OFF_MM     33480001     // 24 uints (12 min-mapped, 12 max-mapped)

typedef __attribute__((ext_vector_type(16))) __bf16 v16bf;
typedef __attribute__((ext_vector_type(8)))  __bf16 v8bf;
typedef __attribute__((ext_vector_type(8)))  float  v8f;

// monotonic float<->uint mapping for atomic min/max
__device__ __forceinline__ unsigned fmap(float f) {
    unsigned u = __float_as_uint(f);
    return (u & 0x80000000u) ? ~u : (u | 0x80000000u);
}
__device__ __forceinline__ float funmap(unsigned m) {
    unsigned u = (m & 0x80000000u) ? (m & 0x7fffffffu) : ~m;
    return __uint_as_float(u);
}
__device__ __forceinline__ float sigm(float x) { return 1.f / (1.f + expf(-x)); }

// image i (0..14) -> sample index in bevs (BT=3,N=5)
__device__ __forceinline__ int bev_sample_for_img(int i) {
    if (i < 3) return i * 5;                 // tg: (bt, n=0)
    int s = i - 3;                           // nb: s = bt*4 + (n-1)
    return (s >> 2) * 5 + 1 + (s & 3);
}

// ---------------- init ----------------
__global__ void k_init(float* ws) {
    unsigned* mm = (unsigned*)&ws[OFF_MM];
    int t = threadIdx.x;
    if (t == 0) ws[OFF_BW] = 0.f;
    if (t < 12)            mm[t] = 0xFFFFFFFFu;  // min slots (largest mapped)
    if (t >= 12 && t < 24) mm[t] = 0u;           // max slots (smallest mapped)
}

// ---------------- compress conv 3x3 (64->1) + relu ----------------
__global__ void k_compress(const float* __restrict__ bevs,
                           const float* __restrict__ cw,
                           const float* __restrict__ cb,
                           float* __restrict__ ws) {
    int gid = blockIdx.x * blockDim.x + threadIdx.x;
    if (gid >= NIMG * HWp) return;
    int i = gid / HWp, p = gid - i * HWp;
    int y = p / Wd, x = p - y * Wd;
    const float* base = bevs + (long)bev_sample_for_img(i) * Cc * HWp;
    float acc = cb[0];
    for (int c = 0; c < Cc; ++c) {
        const float* ch = base + c * HWp;
        const float* wk = cw + c * 9;
        #pragma unroll
        for (int dy = -1; dy <= 1; ++dy) {
            int yy = y + dy; if (yy < 0 || yy >= Hd) continue;
            #pragma unroll
            for (int dx = -1; dx <= 1; ++dx) {
                int xx = x + dx; if (xx < 0 || xx >= Wd) continue;
                acc += ch[yy * Wd + xx] * wk[(dy + 1) * 3 + (dx + 1)];
            }
        }
    }
    ws[OFF_COMP + gid] = fmaxf(acc, 0.f);
}

// ---------------- entropy = mean_{3x3, zero-pad} sigmoid(nbr - center) ----------------
__global__ void k_entropy(float* __restrict__ ws) {
    int gid = blockIdx.x * blockDim.x + threadIdx.x;
    if (gid >= NIMG * HWp) return;
    int i = gid / HWp, p = gid - i * HWp;
    int y = p / Wd, x = p - y * Wd;
    const float* comp = ws + OFF_COMP + i * HWp;
    float xc = comp[p];
    float s = 0.f;
    #pragma unroll
    for (int dy = -1; dy <= 1; ++dy)
        #pragma unroll
        for (int dx = -1; dx <= 1; ++dx) {
            int yy = y + dy, xx = x + dx;
            float v = (yy >= 0 && yy < Hd && xx >= 0 && xx < Wd) ? comp[yy * Wd + xx] : 0.f;
            s += sigm(v - xc);
        }
    ws[OFF_ENT + gid] = s * (1.f / 9.f);
}

// ---------------- mask + bandwidth partial sum ----------------
__global__ void k_mask(float* __restrict__ ws) {
    __shared__ float red[256];
    int gid = blockIdx.x * 256 + threadIdx.x;
    float mval = 0.f;
    if (gid < NS * HWp) {
        int s = gid / HWp, p = gid - s * HWp;
        int bt = s >> 2;
        float me = ws[OFF_ENT + (3 + s) * HWp + p];
        float te = ws[OFF_ENT + bt * HWp + p];
        mval = (me > te) ? 1.f : 0.f;
        ws[OFF_MASK + gid] = mval;
    }
    red[threadIdx.x] = mval;
    __syncthreads();
    for (int o = 128; o > 0; o >>= 1) {
        if (threadIdx.x < o) red[threadIdx.x] += red[threadIdx.x + o];
        __syncthreads();
    }
    if (threadIdx.x == 0) atomicAdd(&ws[OFF_BW], red[0]);
}

// ---------------- den = box3(mask) ----------------
__global__ void k_den(float* __restrict__ ws) {
    int gid = blockIdx.x * 256 + threadIdx.x;
    if (gid >= NS * HWp) return;
    int s = gid / HWp, p = gid - s * HWp;
    int y = p / Wd, x = p - y * Wd;
    const float* mk = ws + OFF_MASK + s * HWp;
    float acc = 0.f;
    #pragma unroll
    for (int dy = -1; dy <= 1; ++dy) {
        int yy = y + dy; if (yy < 0 || yy >= Hd) continue;
        #pragma unroll
        for (int dx = -1; dx <= 1; ++dx) {
            int xx = x + dx; if (xx < 0 || xx >= Wd) continue;
            acc += mk[yy * Wd + xx];
        }
    }
    ws[OFF_DEN + gid] = acc;
}

// ---------------- per-sample min/max of nb ----------------
__global__ void k_minmax(const float* __restrict__ bevs, float* __restrict__ ws) {
    int s = blockIdx.y;                       // 0..11
    int bt = s >> 2, n = 1 + (s & 3);
    const float* base = bevs + (long)(bt * 5 + n) * Cc * HWp;
    int start = blockIdx.x * 25600;           // 100 blocks * 25600 = Cc*HW
    float mn = 3.4e38f, mx = -3.4e38f;
    for (int j = 0; j < 100; ++j) {
        float v = base[start + j * 256 + threadIdx.x];
        mn = fminf(mn, v);
        mx = fmaxf(mx, v);
    }
    __shared__ float rmn[256], rmx[256];
    rmn[threadIdx.x] = mn; rmx[threadIdx.x] = mx;
    __syncthreads();
    for (int o = 128; o > 0; o >>= 1) {
        if (threadIdx.x < o) {
            rmn[threadIdx.x] = fminf(rmn[threadIdx.x], rmn[threadIdx.x + o]);
            rmx[threadIdx.x] = fmaxf(rmx[threadIdx.x], rmx[threadIdx.x + o]);
        }
        __syncthreads();
    }
    if (threadIdx.x == 0) {
        unsigned* mm = (unsigned*)&ws[OFF_MM];
        atomicMin(&mm[s],      fmap(rmn[0]));
        atomicMax(&mm[s + 12], fmap(rmx[0]));
    }
}

// ---------------- warp = m*nb + (1-m)*box3(nb*m)/(box3(m)+eps) ----------------
__global__ void k_warp(const float* __restrict__ bevs, float* __restrict__ ws) {
    int gid = blockIdx.x * 256 + threadIdx.x;        // over 12*64*HW = 30,720,000
    if (gid >= NS * Cc * HWp) return;
    int p = gid % HWp;
    int t = gid / HWp;
    int c = t % Cc;
    int s = t / Cc;
    int bt = s >> 2, n = 1 + (s & 3);
    const float* nb = bevs + ((long)(bt * 5 + n) * Cc + c) * HWp;
    const float* mk = ws + OFF_MASK + s * HWp;
    int y = p / Wd, x = p - y * Wd;
    float num = 0.f;
    #pragma unroll
    for (int dy = -1; dy <= 1; ++dy) {
        int yy = y + dy; if (yy < 0 || yy >= Hd) continue;
        #pragma unroll
        for (int dx = -1; dx <= 1; ++dx) {
            int xx = x + dx; if (xx < 0 || xx >= Wd) continue;
            int q = yy * Wd + xx;
            num += nb[q] * mk[q];
        }
    }
    float den = ws[OFF_DEN + s * HWp + p];
    float m = mk[p];
    float v = nb[p];
    float inter = num / (den + EPSf);
    float w = m * v + (1.f - m) * inter;
    const unsigned* mm = (const unsigned*)&ws[OFF_MM];
    float mnv = funmap(mm[s]), mxv = funmap(mm[s + 12]);
    if (mnv + mxv == 0.f) w = v;
    ws[OFF_WARP + gid] = w;
}

// ---------------- pwf: fused 128->128->32->8->1 per-pixel MLP via WMMA ----------------
// Block = 256 threads = 8 wave32; 128-pixel tile of one image per block.
// LDS: s_x (128x128 bf16 activations, reused across layers), s_w (weights).
__global__ __launch_bounds__(256) void k_pwf(
    const float* __restrict__ bevs,
    const float* __restrict__ w1, const float* __restrict__ b1,
    const float* __restrict__ w2, const float* __restrict__ b2,
    const float* __restrict__ w3, const float* __restrict__ b3,
    const float* __restrict__ w4, const float* __restrict__ b4,
    float* __restrict__ ws) {
    __shared__ __align__(16) __bf16 s_x[128 * 128];
    __shared__ __align__(16) __bf16 s_w[128 * 128];

    const int tid  = threadIdx.x;
    const int img  = blockIdx.y;            // 0..14
    const int bt   = img / 5, n = img % 5;
    const int pix0 = blockIdx.x * 128;

    const float* tg   = bevs + (long)bt * 5 * Cc * HWp;   // sample (bt, n=0)
    const float* warp = ws + OFF_WARP;

    // ---- stage X (128 pixels x 128 ch: [tg | feat_n]) and W1 into LDS ----
    for (int j = 0; j < 64; ++j) {
        int lin = j * 256 + tid;            // 0..16383
        int r = lin & 127, c = lin >> 7;    // consecutive tid -> consecutive pixels
        int p = pix0 + r;
        float v = 0.f;
        if (p < HWp) {
            if (c < 64) v = tg[c * HWp + p];
            else {
                int cc = c - 64;
                if (n == 0) v = tg[cc * HWp + p];
                else        v = warp[((bt * 4 + (n - 1)) * Cc + cc) * HWp + p];
            }
        }
        s_x[r * 128 + c] = (__bf16)v;
        s_w[lin] = (__bf16)w1[lin];         // W1 is (128 out, 128 in) row-major
    }
    __syncthreads();

    const int lane = tid & 31;
    const int wid  = tid >> 5;              // wave 0..7
    const int half = lane >> 4;             // ISA lane-half
    const int rr   = lane & 15;
    const int prow = wid * 16 + rr;         // this lane's A-matrix pixel row

    // A-fragment builder: 16x32 bf16, k = kbase + 16*(e>>3) + 8*half + (e&7)
    #define BUILD_A(dst, kbase)                                                   \
        {                                                                         \
            const __bf16* row_ = &s_x[prow * 128];                                \
            v8bf lo_ = *(const v8bf*)(row_ + (kbase) + 8 * half);                 \
            v8bf hi_ = *(const v8bf*)(row_ + (kbase) + 8 * half + 16);            \
            v16bf t_;                                                             \
            _Pragma("unroll")                                                     \
            for (int e_ = 0; e_ < 8; ++e_) { t_[e_] = lo_[e_]; t_[e_ + 8] = hi_[e_]; } \
            (dst) = t_;                                                           \
        }
    // B-fragment builder from weight row wrow_ptr: k = 16*(e>>3) + 8*half + (e&7)
    #define BUILD_B(dst, wrow_ptr)                                                \
        {                                                                         \
            v8bf lo_ = *(const v8bf*)((wrow_ptr) + 8 * half);                     \
            v8bf hi_ = *(const v8bf*)((wrow_ptr) + 8 * half + 16);                \
            v16bf t_;                                                             \
            _Pragma("unroll")                                                     \
            for (int e_ = 0; e_ < 8; ++e_) { t_[e_] = lo_[e_]; t_[e_ + 8] = hi_[e_]; } \
            (dst) = t_;                                                           \
        }

    // ---- layer 1: K=128 -> 128 out ----
    v16bf a[4];
    #pragma unroll
    for (int kk = 0; kk < 4; ++kk) BUILD_A(a[kk], kk * 32);

    #pragma unroll
    for (int tcol = 0; tcol < 8; ++tcol) {
        v8f acc = {};
        #pragma unroll
        for (int kk = 0; kk < 4; ++kk) {
            v16bf b;
            BUILD_B(b, &s_w[(tcol * 16 + rr) * 128 + kk * 32]);
            acc = __builtin_amdgcn_wmma_f32_16x16x32_bf16(
                false, a[kk], false, b, (short)0, acc, false, false);
        }
        float bias = b1[tcol * 16 + rr];
        #pragma unroll
        for (int g = 0; g < 8; ++g) {   // D: row = g + 8*half, col = rr
            float o = fmaxf(acc[g] + bias, 0.f);
            s_x[(wid * 16 + g + 8 * half) * 128 + tcol * 16 + rr] = (__bf16)o;
        }
    }
    __syncthreads();   // all waves done with W1 before overwrite

    // stage W2 (32x128) and W3 (8x32; pad rows 8..15 not zeroed, hold stale W1) into LDS
    for (int j = 0; j < 16; ++j) {
        int lin = j * 256 + tid;            // 0..4095
        s_w[lin] = (__bf16)w2[lin];
    }
    if (tid < 512) {
        int o = tid >> 5, k = tid & 31;
        s_w[4096 + tid] = (__bf16)((o < 8) ? w3[o * 32 + k] : 0.f);
    }
    __syncthreads();

    // ---- layer 2: K=128 -> 32 out ----
    v16bf a2[4];
    #pragma unroll
    for (int kk = 0; kk < 4; ++kk) BUILD_A(a2[kk], kk * 32);

    #pragma unroll
    for (int tcol = 0; tcol < 2; ++tcol) {
        v8f acc = {};
        #pragma unroll
        for (int kk = 0; kk < 4; ++kk) {
            v16bf b;
            BUILD_B(b, &s_w[(tcol * 16 + rr) * 128 + kk * 32]);
            acc = __builtin_amdgcn_wmma_f32_16x16x32_bf16(
                false, a2[kk], false, b, (short)0, acc, false, false);
        }
        float bias = b2[tcol * 16 + rr];
        #pragma unroll
        for (int g = 0; g < 8; ++g) {
            float o = fmaxf(acc[g] + bias, 0.f);
            s_x[(wid * 16 + g + 8 * half) * 128 + tcol * 16 + rr] = (__bf16)o;
        }
    }

    // ---- layer 3: K=32 -> 8 out (tile padded to 16) ----
    {
        v16bf a3;
        BUILD_A(a3, 0);
        v16bf b;
        BUILD_B(b, &s_w[4096 + rr * 32]);
        v8f acc = {};
        acc = __builtin_amdgcn_wmma_f32_16x16x32_bf16(
            false, a3, false, b, (short)0, acc, false, false);
        if (rr < 8) {
            float bias = b3[rr];
            #pragma unroll
            for (int g = 0; g < 8; ++g) {
                float o = fmaxf(acc[g] + bias, 0.f);
                s_x[(wid * 16 + g + 8 * half) * 128 + rr] = (__bf16)o;
            }
        }
    }
    __syncthreads();

    // ---- layer 4: 8 -> 1, plain dot ----
    if (tid < 128) {
        int p = pix0 + tid;
        if (p < HWp) {
            float acc = b4[0];
            #pragma unroll
            for (int j = 0; j < 8; ++j)
                acc += (float)s_x[tid * 128 + j] * w4[j];
            ws[OFF_LOGIT + img * HWp + p] = fmaxf(acc, 0.f);
        }
    }
    #undef BUILD_A
    #undef BUILD_B
}

// ---------------- softmax-over-N weighting + fuse, write bandwidth ----------------
__global__ void k_fuse(const float* __restrict__ bevs,
                       const float* __restrict__ ws,
                       float* __restrict__ out) {
    int gid = blockIdx.x * 256 + threadIdx.x;        // over 3*64*HW
    if (gid < 3 * Cc * HWp) {
        int p  = gid % HWp;
        int t  = gid / HWp;
        int c  = t % Cc;
        int bt = t / Cc;
        const float* logit = ws + OFF_LOGIT + (bt * 5) * HWp;
        float wv[5], wsum = 0.f;
        #pragma unroll
        for (int n = 0; n < 5; ++n) { wv[n] = expf(logit[n * HWp + p]); wsum += wv[n]; }
        float inv = 1.f / wsum;
        float fsum = wv[0] * bevs[((long)(bt * 5) * Cc + c) * HWp + p];  // feat_0 = tg
        #pragma unroll
        for (int n = 1; n < 5; ++n)
            fsum += wv[n] * ws[OFF_WARP + ((bt * 4 + n - 1) * Cc + c) * HWp + p];
        out[gid] = fsum * inv;
    }
    if (blockIdx.x == 0 && threadIdx.x == 0)
        out[3 * Cc * HWp] = ws[OFF_BW] * (1.f / (float)(NS * HWp));
}

// ---------------- launch ----------------
extern "C" void kernel_launch(void* const* d_in, const int* in_sizes, int n_in,
                              void* d_out, int out_size, void* d_ws, size_t ws_size,
                              hipStream_t stream) {
    const float* bevs = (const float*)d_in[0];
    const float* cw   = (const float*)d_in[1];
    const float* cb   = (const float*)d_in[2];
    const float* w1   = (const float*)d_in[3];
    const float* b1   = (const float*)d_in[4];
    const float* w2   = (const float*)d_in[5];
    const float* b2   = (const float*)d_in[6];
    const float* w3   = (const float*)d_in[7];
    const float* b3   = (const float*)d_in[8];
    const float* w4   = (const float*)d_in[9];
    const float* b4   = (const float*)d_in[10];
    float* out = (float*)d_out;
    float* ws  = (float*)d_ws;     // needs ~134 MB

    k_init<<<1, 32, 0, stream>>>(ws);
    k_compress<<<(NIMG * HWp + 255) / 256, 256, 0, stream>>>(bevs, cw, cb, ws);
    k_entropy <<<(NIMG * HWp + 255) / 256, 256, 0, stream>>>(ws);
    k_mask    <<<(NS * HWp + 255) / 256, 256, 0, stream>>>(ws);
    k_den     <<<(NS * HWp + 255) / 256, 256, 0, stream>>>(ws);
    k_minmax  <<<dim3(100, 12), 256, 0, stream>>>(bevs, ws);
    k_warp    <<<(NS * Cc * HWp + 255) / 256, 256, 0, stream>>>(bevs, ws);
    k_pwf     <<<dim3((HWp + 127) / 128, NIMG), 256, 0, stream>>>(
                  bevs, w1, b1, w2, b2, w3, b3, w4, b4, ws);
    k_fuse    <<<(3 * Cc * HWp + 255) / 256, 256, 0, stream>>>(bevs, ws, out);
}
